// SimpleLSTM_29669634081315
// MI455X (gfx1250) — matrix operand order repaired and checked by reading, not compiled, not run on hardware
//
#include <hip/hip_runtime.h>
#include <hip/hip_bf16.h>

typedef __attribute__((ext_vector_type(2)))  float  v2f;
typedef __attribute__((ext_vector_type(8)))  float  v8f;
typedef __attribute__((ext_vector_type(2)))  __bf16 v2bf;
typedef __attribute__((ext_vector_type(16))) __bf16 v16bf;

// ---- f32 -> packed bf16 pair: hardware v_cvt_pk_bf16_f32 when available ----
__device__ __forceinline__ v2bf cvt2bf(float a, float b) {
#if __has_builtin(__builtin_amdgcn_cvt_pk_bf16_f32)
    return __builtin_amdgcn_cvt_pk_bf16_f32(a, b);
#else
    unsigned ua = __float_as_uint(a), ub = __float_as_uint(b);
    ua = (ua + 0x7FFFu + ((ua >> 16) & 1u)) >> 16;
    ub = (ub + 0x7FFFu + ((ub >> 16) & 1u)) >> 16;
    union { unsigned u; v2bf v; } c;
    c.u = (ua & 0xFFFFu) | (ub << 16);
    return c.v;
#endif
}

// ---- transcendentals: v_tanh_f32 (CDNA5 TRANS op) when available ----
__device__ __forceinline__ float tanh_t(float x) {
#if __has_builtin(__builtin_amdgcn_tanhf)
    return __builtin_amdgcn_tanhf(x);
#else
    return 1.0f - 2.0f * __builtin_amdgcn_rcpf(1.0f + __builtin_amdgcn_exp2f(2.8853900817779268f * x));
#endif
}
// sigmoid of z where the *pre-activation was already halved* in the weights:
// sigmoid(z) = 0.5*tanh(z/2) + 0.5, and x == z/2 here.
__device__ __forceinline__ float sigmoid_half(float x) {
#if __has_builtin(__builtin_amdgcn_tanhf)
    return fmaf(0.5f, __builtin_amdgcn_tanhf(x), 0.5f);
#else
    return __builtin_amdgcn_rcpf(1.0f + __builtin_amdgcn_exp2f(-2.8853900817779268f * x));
#endif
}

union AFrag  { uint4 q[2]; v16bf v; };
union BFrag  { unsigned u[8]; v16bf v; };

// B=4096, T=512, I=3, H=32, 4H=128.  One wave of 32 lanes owns 16 batch rows.
// grid = 64 blocks * 128 threads (4 waves) -> 256 waves * 16 rows = 4096 rows.
__global__ __launch_bounds__(128)
void lstm_wmma_kernel(const float* __restrict__ x,
                      const float* __restrict__ w_ih,
                      const float* __restrict__ w_hh,
                      const float* __restrict__ b_ih,
                      const float* __restrict__ b_hh,
                      const float* __restrict__ w_fc,
                      const float* __restrict__ b_fc,
                      float* __restrict__ out) {
    constexpr int T = 512, I = 3, H = 32;
    constexpr int LP = 40;           // bf16 row pitch: 16B-aligned, bank-spread
    const int lane  = threadIdx.x & 31;
    const int wave  = threadIdx.x >> 5;
    const int half  = lane >> 4;     // 0: lanes 0-15, 1: lanes 16-31
    const int l16   = lane & 15;
    const int base_row = (blockIdx.x * 4 + wave) * 16;

    // Wave-private h tile in bf16, row-major [16 rows][32 cols] (pitch 40)
    __shared__ __bf16 lds_h[4][16][LP];
    __bf16 (*hb)[LP] = lds_h[wave];

    // Gate tiles: i={0,1} f={2,3} g={4,5} o={6,7}.  Sigmoid gates (i,f,o) get
    // their weights+bias pre-scaled by 0.5 (exact in bf16: power of two), so
    // sigmoid(z) = fma(0.5, tanh(acc), 0.5) with no per-element pre-multiply.

    // ---- B fragments for w_hh.T (bf16, 8 N-tiles of 16 gate columns) ----
    // Dense 32x16 B layout (N=lane%16): VGPR j<4 packs K = 8*half + 2j (lo), +1 (hi);
    // j>=4 packs K = 16 + 8*half + 2(j-4) (lo), +1 (hi).
    v16bf Bh[8];
    #pragma unroll
    for (int n = 0; n < 8; ++n) {
        const float s = (n == 4 || n == 5) ? 1.0f : 0.5f;
        BFrag bb;
        const float* wr = w_hh + (16 * n + l16) * H + 8 * half;  // w_hh[gate][k]
        #pragma unroll
        for (int j = 0; j < 8; ++j) {
            v2bf p = cvt2bf(s * wr[2 * j + (j >= 4 ? 8 : 0)], s * wr[2 * j + (j >= 4 ? 8 : 0) + 1]);
            union { v2bf v; unsigned u; } c; c.v = p;
            bb.u[j] = c.u;
        }
        Bh[n] = bb.v;
    }

    // ---- B fragments for [w_ih.T ; bias] (f32 4x16 per N-tile) ----
    // B 4x16 layout: VGPR0 = {K=0 lanes0-15, K=2 lanes16-31}, VGPR1 = {K=1, K=3}.
    // K=0..2 are w_ih columns, K=3 row is (b_ih+b_hh) multiplied by constant A=1.
    v2f Bx[8];
    #pragma unroll
    for (int n = 0; n < 8; ++n) {
        const float s = (n == 4 || n == 5) ? 1.0f : 0.5f;
        int g = 16 * n + l16;
        v2f b;
        if (half == 0) { b[0] = s * w_ih[g * I + 0]; b[1] = s * w_ih[g * I + 1]; }
        else           { b[0] = s * w_ih[g * I + 2]; b[1] = s * (b_ih[g] + b_hh[g]); }
        Bx[n] = b;
    }

    // ---- zero-init h in LDS (h0 = 0), c = 0 ----
    #pragma unroll
    for (int u = 0; u < 2; ++u)
        #pragma unroll
        for (int r = 0; r < 8; ++r)
            hb[r + 8 * half][16 * u + l16] = (__bf16)0.0f;
    asm volatile("s_wait_dscnt 0" ::: "memory");

    v8f c0 = {}, c1 = {};
    const v8f zeroC = {};
    const float* xrow = x + (size_t)(base_row + l16) * T * I;  // this lane's batch row
    const int k0 = 8 * half;

    #pragma unroll 1
    for (int t = 0; t < T; ++t) {
        // ---- A fragment for [x | 1] (f32 16x4): lane m<16: K0=x0,K1=x1; m>=16: K2=x2,K3=1
        // (upper half loads a safe duplicate and selects 1.0f -> no EXEC toggling)
        const int o0 = t * I + 2 * half;       // x0 (half0) / x2 (half1)
        const int o1 = t * I + 1 + half;       // x1 (half0) / x2 again (half1, in-bounds)
        v2f ax;
        ax[0] = xrow[o0];
        float xv = xrow[o1];
        ax[1] = half ? 1.0f : xv;
        __builtin_prefetch(xrow + t * I + 48, 0, 3);   // stream x ~16 steps ahead

        // ---- A fragment for h (bf16 16x32) straight from LDS: two ds_load_b128 ----
        // lane l: row = l%16; VGPR j<4: K = 8*half + 2j ; j>=4: K = 16 + 8*half + 2(j-4)
        asm volatile("s_wait_dscnt 0" ::: "memory");   // previous step's h store visible
        AFrag ah;
        ah.q[0] = *(const uint4*)&hb[l16][k0];
        ah.q[1] = *(const uint4*)&hb[l16][16 + k0];

        // ---- gates = [x|1] @ [w_ih.T;bias]  +  h @ w_hh.T   (16 x 128) ----
        v8f acc[8];
        #pragma unroll
        for (int n = 0; n < 8; ++n) {
            acc[n] = __builtin_amdgcn_wmma_f32_16x16x4_f32(
                         false, ax, false, Bx[n], (short)0, zeroC, false, false);
            acc[n] = __builtin_amdgcn_wmma_f32_16x16x32_bf16(
                         false, ah.v, false, Bh[n], (short)0, acc[n], false, false);
        }

        // ---- elementwise LSTM cell (PyTorch gate order i,f,g,o across N-tiles) ----
        // ---- then write h back as bf16 (one v_cvt_pk + two b16 stores per r) ----
        #pragma unroll
        for (int r = 0; r < 8; ++r) {
            float i0 = sigmoid_half(acc[0][r]);
            float f0 = sigmoid_half(acc[2][r]);
            float g0 = tanh_t      (acc[4][r]);
            float o0g= sigmoid_half(acc[6][r]);
            float cv0 = f0 * c0[r] + i0 * g0;
            c0[r] = cv0;
            float h0 = o0g * tanh_t(cv0);

            float i1 = sigmoid_half(acc[1][r]);
            float f1 = sigmoid_half(acc[3][r]);
            float g1 = tanh_t      (acc[5][r]);
            float o1g= sigmoid_half(acc[7][r]);
            float cv1 = f1 * c1[r] + i1 * g1;
            c1[r] = cv1;
            float h1 = o1g * tanh_t(cv1);

            v2bf pk = cvt2bf(h0, h1);          // cols l16 and 16+l16 of row r+8*half
            hb[r + 8 * half][l16]      = pk[0];
            hb[r + 8 * half][16 + l16] = pk[1];
        }
    }

    // ---- final FC: out[row] = h_T[row] . w_fc + b_fc ----
    asm volatile("s_wait_dscnt 0" ::: "memory");
    if (lane < 16) {
        float p = b_fc[0];
        #pragma unroll
        for (int k = 0; k < H; ++k) p += (float)hb[lane][k] * w_fc[k];
        out[base_row + lane] = p;
    }
}

extern "C" void kernel_launch(void* const* d_in, const int* in_sizes, int n_in,
                              void* d_out, int out_size, void* d_ws, size_t ws_size,
                              hipStream_t stream) {
    const float* x    = (const float*)d_in[0];
    const float* w_ih = (const float*)d_in[1];
    const float* w_hh = (const float*)d_in[2];
    const float* b_ih = (const float*)d_in[3];
    const float* b_hh = (const float*)d_in[4];
    const float* w_fc = (const float*)d_in[5];
    const float* b_fc = (const float*)d_in[6];

    dim3 grid(64), block(128);  // 4 waves/block, 16 batch rows per wave -> 4096 rows
    lstm_wmma_kernel<<<grid, block, 0, stream>>>(x, w_ih, w_hh, b_ih, b_hh,
                                                 w_fc, b_fc, (float*)d_out);
}
